// GraphConvolutionalNetwork_17540646437552
// MI455X (gfx1250) — compile-verified, matrix operand order checked by reading, not run on hardware
//
#include <hip/hip_runtime.h>
#include <hip/hip_bf16.h>
#include <math.h>

typedef __attribute__((ext_vector_type(16))) _Float16 v16h;
typedef __attribute__((ext_vector_type(8)))  _Float16 v8h;
typedef __attribute__((ext_vector_type(8)))  float    v8f;

// ---------------------------------------------------------------------------
// Elementwise helpers
// ---------------------------------------------------------------------------
__global__ void fill_zero_u32(unsigned int* p, long long n) {
    long long i = (long long)blockIdx.x * blockDim.x + threadIdx.x;
    if (i < n) p[i] = 0u;
}

__global__ void f32_to_f16_kernel(const float* __restrict__ in, _Float16* __restrict__ out, long long n) {
    long long i = (long long)blockIdx.x * blockDim.x + threadIdx.x;
    if (i < n) out[i] = (_Float16)in[i];
}

// W[K][Ho] f32 (row-major, as in reference) -> Wt[Ho][K] f16
__global__ void transpose_to_f16_kernel(const float* __restrict__ W, _Float16* __restrict__ Wt, int K, int Ho) {
    long long i = (long long)blockIdx.x * blockDim.x + threadIdx.x;
    if (i >= (long long)K * Ho) return;
    int k = (int)(i / Ho), n = (int)(i % Ho);
    Wt[(long long)n * K + k] = (_Float16)W[i];
}

// ---------------------------------------------------------------------------
// Degree / normalization
// ---------------------------------------------------------------------------
__global__ void count_deg_kernel(const long long* __restrict__ dst, int E, int* __restrict__ cnt) {
    int e = blockIdx.x * blockDim.x + threadIdx.x;
    if (e < E) atomicAdd(&cnt[(int)dst[e]], 1);
}

__global__ void dinv_kernel(const int* __restrict__ cnt, float* __restrict__ dinv, int N) {
    int n = blockIdx.x * blockDim.x + threadIdx.x;
    if (n < N) dinv[n] = rsqrtf((float)(cnt[n] + 1));   // +1 self-loop, always > 0
}

// out[n][f] = h[n][f] * dinv[n]^2   (self-loop contribution; also initializes out)
__global__ void agg_init_kernel(const float* __restrict__ h, const float* __restrict__ dinv,
                                float* __restrict__ out, int N, int H) {
    long long i = (long long)blockIdx.x * blockDim.x + threadIdx.x;
    if (i >= (long long)N * H) return;
    int n = (int)(i / H);
    float d = dinv[n];
    out[i] = h[i] * d * d;
}

// one wave per edge: out[dst] += h[src] * dinv[src]*dinv[dst]
__global__ __launch_bounds__(256) void edge_scatter_kernel(
    const float* __restrict__ h, const float* __restrict__ dinv,
    const long long* __restrict__ src, const long long* __restrict__ dst,
    int E, int H, float* __restrict__ out) {
    int gwave = (int)(((long long)blockIdx.x * blockDim.x + threadIdx.x) >> 5);
    int lane = threadIdx.x & 31;
    if (gwave >= E) return;
    int s = (int)src[gwave];
    int d = (int)dst[gwave];
    float w = dinv[s] * dinv[d];
    const float* hs = h + (long long)s * H;
    float* od = out + (long long)d * H;
    for (int f = lane; f < H; f += 32)
        atomicAdd(&od[f], hs[f] * w);
}

// x1 = relu(a + b1[col]); write f32 and f16 copies
__global__ void relu_bias_dual_kernel(const float* __restrict__ a, const float* __restrict__ bias,
                                      float* __restrict__ x, _Float16* __restrict__ xh, int N, int H) {
    long long i = (long long)blockIdx.x * blockDim.x + threadIdx.x;
    if (i >= (long long)N * H) return;
    int f = (int)(i % H);
    float v = a[i] + bias[f];
    v = fmaxf(v, 0.0f);
    x[i] = v;
    xh[i] = (_Float16)v;
}

// acc += a2 + b2[col] + x1   (residual + snapshot accumulation)
__global__ void x2_accumulate_kernel(const float* __restrict__ a2, const float* __restrict__ bias,
                                     const float* __restrict__ x1, float* __restrict__ acc, int N, int H) {
    long long i = (long long)blockIdx.x * blockDim.x + threadIdx.x;
    if (i >= (long long)N * H) return;
    int f = (int)(i % H);
    acc[i] += a2[i] + bias[f] + x1[i];
}

// ph[p][f] = f16( acc[post_idx[p]][f] * invS )
__global__ void post_gather_kernel(const float* __restrict__ acc, const long long* __restrict__ post_idx,
                                   _Float16* __restrict__ ph, int P, int H, float invS) {
    long long i = (long long)blockIdx.x * blockDim.x + threadIdx.x;
    if (i >= (long long)P * H) return;
    int p = (int)(i / H), f = (int)(i % H);
    ph[i] = (_Float16)(acc[(long long)post_idx[p] * H + f] * invS);
}

// out[p] = sigmoid( dot(z1[p], Wc2) + bc2 )
__global__ void final_logit_kernel(const float* __restrict__ z1, const float* __restrict__ Wc2,
                                   const float* __restrict__ bc2, float* __restrict__ out, int P, int Hh) {
    int p = blockIdx.x * blockDim.x + threadIdx.x;
    if (p >= P) return;
    float s = bc2[0];
    const float* z = z1 + (long long)p * Hh;
    for (int i = 0; i < Hh; ++i) s += z[i] * Wc2[i];
    out[p] = 1.0f / (1.0f + expf(-s));
}

// ---------------------------------------------------------------------------
// WMMA GEMM: C[M][Ncol] = A[M][K](f16) * Bt[Ncol][K](f16)^T  (+bias, +relu)
//
// Block = 256 threads = 8 waves. Each wave computes FOUR 16x16 row tiles of
// one 16-column tile (wave: 64 rows x 16 cols; block: 512 rows x 16 cols).
// The shared 16xK B tile is staged once per block into LDS (row stride K+8
// halves -> 16B-aligned rows, bank-rotated so lane b128 reads are
// conflict-free) and re-read with ds_load_b128. Per K-step: 2 LDS b128 (B) +
// 8 global b128 (A) + 4 independent v_wmma_f32_16x16x32_f16.
//
// Fragment layout per CDNA5 16-bit A/B spec:
//   halves 0..7  <- K = k0 + 8*(lane>=16) + 0..7
//   halves 8..15 <- K = k0 + 16 + 8*(lane>=16) + 0..7
//
// M (grid rows) must be a multiple of 512: callers pad row dimensions in the
// workspace; pad rows are garbage-in/garbage-out and never consumed. No
// branches in the body -> EXEC all-ones for every WMMA.
// ---------------------------------------------------------------------------
#define GEMM_MAXK 256

__global__ __launch_bounds__(256) void wmma_gemm_kernel(
    const _Float16* __restrict__ A, const _Float16* __restrict__ Bt,
    float* __restrict__ C, const float* __restrict__ bias,
    int K, int Ncol, int relu) {
    __shared__ _Float16 sBt[16 * (GEMM_MAXK + 8)];
    const int tid     = threadIdx.x;
    const int lane    = tid & 31;
    const int wave    = tid >> 5;
    const int colTile = blockIdx.x;
    const int ldb     = K + 8;                 // LDS row stride in halves

    // ---- stage 16 x K B tile into LDS (16B chunks) ----
    {
        const int chunksPerRow = K >> 3;       // K/8 halves per 16B chunk
        const int total = 16 * chunksPerRow;
        for (int idx = tid; idx < total; idx += 256) {
            int row = idx / chunksPerRow;
            int c   = idx - row * chunksPerRow;
            *(v8h*)&sBt[row * ldb + c * 8] =
                *(const v8h*)&Bt[(long long)(colTile * 16 + row) * K + c * 8];
        }
    }
    __syncthreads();

    const int l15  = lane & 15;
    const int half = lane >> 4;
    const int rowBase = blockIdx.y * 512 + wave * 64;   // 4 row tiles per wave
    const _Float16* aPtr = A + (long long)(rowBase + l15) * K + half * 8;
    const _Float16* bLds = &sBt[l15 * ldb + half * 8];
    const long long aRowTileStride = (long long)16 * K; // halves between row tiles

    v8f acc[4] = {v8f{}, v8f{}, v8f{}, v8f{}};
    for (int k0 = 0; k0 < K; k0 += 32) {
        v16h bf;
        ((v8h*)&bf)[0] = *(const v8h*)(bLds + k0);
        ((v8h*)&bf)[1] = *(const v8h*)(bLds + k0 + 16);
        #pragma unroll
        for (int r = 0; r < 4; ++r) {
            v16h af;
            const _Float16* ap = aPtr + r * aRowTileStride + k0;
            ((v8h*)&af)[0] = *(const v8h*)(ap);
            ((v8h*)&af)[1] = *(const v8h*)(ap + 16);
            acc[r] = __builtin_amdgcn_wmma_f32_16x16x32_f16(
                /*neg_a=*/false, af, /*neg_b=*/false, bf,
                /*c_mod=*/(short)0, acc[r], /*reuse_a=*/false, /*reuse_b=*/false);
        }
    }

    const int col = colTile * 16 + l15;
    const float bv = bias ? bias[col] : 0.0f;
    #pragma unroll
    for (int r = 0; r < 4; ++r) {
        const int mb = rowBase + r * 16 + half * 8;   // C/D layout: VGPR j -> M = 8*half + j
        #pragma unroll
        for (int j = 0; j < 8; ++j) {
            float v = acc[r][j] + bv;
            if (relu) v = fmaxf(v, 0.0f);
            C[(long long)(mb + j) * Ncol + col] = v;
        }
    }
}

// ---------------------------------------------------------------------------
// Host-side orchestration
// ---------------------------------------------------------------------------
static inline long long cdiv_ll(long long a, long long b) { return (a + b - 1) / b; }
static inline long long round_up_ll(long long a, long long b) { return cdiv_ll(a, b) * b; }

extern "C" void kernel_launch(void* const* d_in, const int* in_sizes, int n_in,
                              void* d_out, int out_size, void* d_ws, size_t ws_size,
                              hipStream_t stream) {
    const float*     X   = (const float*)d_in[0];      // [N, D_IN]
    const long long* EI  = (const long long*)d_in[1];  // [S, 2, E]
    const long long* PIX = (const long long*)d_in[2];  // [P]
    const float*     W1  = (const float*)d_in[3];
    const float*     b1  = (const float*)d_in[4];
    const float*     W2  = (const float*)d_in[5];
    const float*     b2  = (const float*)d_in[6];
    const float*     Wc1 = (const float*)d_in[7];
    const float*     bc1 = (const float*)d_in[8];
    const float*     Wc2 = (const float*)d_in[9];
    const float*     bc2 = (const float*)d_in[10];
    float*           out = (float*)d_out;

    const int S    = 5;
    const int H    = in_sizes[4];                 // 256
    const int D_IN = in_sizes[3] / H;             // 128
    const int Hc   = in_sizes[8];                 // 128
    const int N    = in_sizes[0] / D_IN;          // 100000
    const int P    = in_sizes[2];                 // 10000
    const int E    = in_sizes[1] / (2 * S);       // 300000

    const long long Npad = round_up_ll(N, 512);   // GEMM row-dim padding
    const long long Ppad = round_up_ll(P, 512);

    // ---- carve workspace (256B aligned regions) ----
    char* base = (char*)d_ws;
    size_t off = 0;
    auto carve = [&](size_t bytes) -> void* {
        void* p = base + off;
        off += (bytes + 255) & ~(size_t)255;
        return p;
    };
    float*    h1    = (float*)carve((size_t)Npad * H * sizeof(float));
    float*    agg   = (float*)carve((size_t)N * H * sizeof(float));
    float*    x1    = (float*)carve((size_t)N * H * sizeof(float));
    float*    h2    = (float*)carve((size_t)Npad * H * sizeof(float));
    float*    acc   = (float*)carve((size_t)N * H * sizeof(float));
    float*    dinv  = (float*)carve((size_t)N * sizeof(float));
    float*    z1    = (float*)carve((size_t)Ppad * Hc * sizeof(float));
    int*      cnt   = (int*)carve((size_t)N * sizeof(int));
    _Float16* Xh    = (_Float16*)carve((size_t)Npad * D_IN * sizeof(_Float16));
    _Float16* x1h   = (_Float16*)carve((size_t)Npad * H * sizeof(_Float16));
    _Float16* ph    = (_Float16*)carve((size_t)Ppad * H * sizeof(_Float16));
    _Float16* W1t   = (_Float16*)carve((size_t)D_IN * H * sizeof(_Float16));
    _Float16* W2t   = (_Float16*)carve((size_t)H * H * sizeof(_Float16));
    _Float16* Wc1t  = (_Float16*)carve((size_t)H * Hc * sizeof(_Float16));
    (void)ws_size;

    const int TB = 256;
    const long long NH = (long long)N * H;

    // ---- one-time prep ----
    // Zero padded f16 A-buffers' tails once per launch (cheap, keeps pad rows finite).
    fill_zero_u32<<<(unsigned)cdiv_ll((long long)Npad * D_IN / 2, TB), TB, 0, stream>>>((unsigned int*)Xh, (long long)Npad * D_IN / 2);
    f32_to_f16_kernel<<<(unsigned)cdiv_ll((long long)N * D_IN, TB), TB, 0, stream>>>(X, Xh, (long long)N * D_IN);
    transpose_to_f16_kernel<<<(unsigned)cdiv_ll((long long)D_IN * H, TB), TB, 0, stream>>>(W1, W1t, D_IN, H);
    transpose_to_f16_kernel<<<(unsigned)cdiv_ll((long long)H * H, TB), TB, 0, stream>>>(W2, W2t, H, H);
    transpose_to_f16_kernel<<<(unsigned)cdiv_ll((long long)H * Hc, TB), TB, 0, stream>>>(Wc1, Wc1t, H, Hc);
    fill_zero_u32<<<(unsigned)cdiv_ll(NH, TB), TB, 0, stream>>>((unsigned int*)acc, NH);
    fill_zero_u32<<<(unsigned)cdiv_ll((long long)Npad * H / 2, TB), TB, 0, stream>>>((unsigned int*)x1h, (long long)Npad * H / 2);
    fill_zero_u32<<<(unsigned)cdiv_ll((long long)Ppad * H / 2, TB), TB, 0, stream>>>((unsigned int*)ph, (long long)Ppad * H / 2);

    // h1 = X @ W1  (shared by all snapshots: node features are time-invariant)
    {
        dim3 grid(H / 16, (unsigned)(Npad / 512));
        wmma_gemm_kernel<<<grid, TB, 0, stream>>>(Xh, W1t, h1, nullptr, D_IN, H, 0);
    }

    // ---- per-snapshot pipeline ----
    for (int s = 0; s < S; ++s) {
        const long long* src = EI + (long long)s * 2 * E;
        const long long* dst = src + E;

        fill_zero_u32<<<(unsigned)cdiv_ll(N, TB), TB, 0, stream>>>((unsigned int*)cnt, N);
        count_deg_kernel<<<(unsigned)cdiv_ll(E, TB), TB, 0, stream>>>(dst, E, cnt);
        dinv_kernel<<<(unsigned)cdiv_ll(N, TB), TB, 0, stream>>>(cnt, dinv, N);

        // layer 0 aggregation of h1
        agg_init_kernel<<<(unsigned)cdiv_ll(NH, TB), TB, 0, stream>>>(h1, dinv, agg, N, H);
        edge_scatter_kernel<<<(unsigned)cdiv_ll((long long)E * 32, TB), TB, 0, stream>>>(h1, dinv, src, dst, E, H, agg);
        relu_bias_dual_kernel<<<(unsigned)cdiv_ll(NH, TB), TB, 0, stream>>>(agg, b1, x1, x1h, N, H);

        // layer 1: h2 = x1 @ W2
        {
            dim3 grid(H / 16, (unsigned)(Npad / 512));
            wmma_gemm_kernel<<<grid, TB, 0, stream>>>(x1h, W2t, h2, nullptr, H, H, 0);
        }
        agg_init_kernel<<<(unsigned)cdiv_ll(NH, TB), TB, 0, stream>>>(h2, dinv, agg, N, H);
        edge_scatter_kernel<<<(unsigned)cdiv_ll((long long)E * 32, TB), TB, 0, stream>>>(h2, dinv, src, dst, E, H, agg);
        x2_accumulate_kernel<<<(unsigned)cdiv_ll(NH, TB), TB, 0, stream>>>(agg, b2, x1, acc, N, H);
    }

    // ---- head ----
    post_gather_kernel<<<(unsigned)cdiv_ll((long long)P * H, TB), TB, 0, stream>>>(acc, PIX, ph, P, H, 1.0f / (float)S);
    {
        dim3 grid(Hc / 16, (unsigned)(Ppad / 512));
        wmma_gemm_kernel<<<grid, TB, 0, stream>>>(ph, Wc1t, z1, bc1, H, Hc, 1);
    }
    final_logit_kernel<<<(unsigned)cdiv_ll(P, TB), TB, 0, stream>>>(z1, Wc2, bc2, out, P, Hc);
}
